// SumformerInnerBlock_34961033789781
// MI455X (gfx1250) — matrix operand order repaired and checked by reading, not compiled, not run on hardware
//
#include <hip/hip_runtime.h>
#include <stdint.h>

// ---------------------------------------------------------------------------
// Sumformer inner block on MI455X (gfx1250): all GEMMs via
// v_wmma_f32_16x16x32_bf16 (wave32, f32 accum), bf16 intermediates,
// pre-transposed bf16 weights, double-buffered LDS staging, optional
// CDNA5 async global->LDS copies (ASYNCcnt).
// ---------------------------------------------------------------------------

typedef __attribute__((ext_vector_type(16))) __bf16 v16bf;
typedef __attribute__((ext_vector_type(8)))  float  v8f;

#define LRELU_SLOPE 0.01f

// ---- CDNA5 async global->LDS support (guarded; falls back to sync copy) ----
#if defined(__has_builtin)
#  if __has_builtin(__builtin_amdgcn_global_load_async_to_lds_b128)
#    define HAVE_ASYNC_LDS 1
#  endif
#endif
#ifndef HAVE_ASYNC_LDS
#  define HAVE_ASYNC_LDS 0
#endif

#if HAVE_ASYNC_LDS
// Param 0 of the builtin is 'int __vector(4) __device__ *' per clang, i.e. a
// vector_size(16) int pointer in addrspace(1); LDS side is addrspace(3).
typedef int v4i_vs __attribute__((vector_size(4 * sizeof(int))));
typedef __attribute__((address_space(1))) v4i_vs gv4;
typedef __attribute__((address_space(3))) v4i_vs lv4;
#endif

__device__ __forceinline__ void wait_async_lds() {
#if HAVE_ASYNC_LDS
#  if __has_builtin(__builtin_amdgcn_s_wait_asynccnt)
  __builtin_amdgcn_s_wait_asynccnt(0);
#  else
  asm volatile("s_wait_asynccnt 0x0" ::: "memory");
#  endif
#endif
}

__device__ __forceinline__ unsigned short f2bf(float f) {
  union { float f; unsigned int u; } c; c.f = f;
  unsigned int u = c.u;
  unsigned int r = (u + 0x7FFFu + ((u >> 16) & 1u)) >> 16;  // RNE
  return (unsigned short)r;
}
__device__ __forceinline__ float bf2f(unsigned short h) {
  union { float f; unsigned int u; } c; c.u = ((unsigned int)h) << 16;
  return c.f;
}
__device__ __forceinline__ float lrelu(float v) {
  return v >= 0.f ? v : LRELU_SLOPE * v;
}

// --------------------------- f32 -> bf16 convert ---------------------------
__global__ void cvt_f32_bf16(const float* __restrict__ in,
                             unsigned short* __restrict__ out, int n) {
  int i = blockIdx.x * blockDim.x + threadIdx.x;
  int stride = gridDim.x * blockDim.x;
  for (; i < n; i += stride) out[i] = f2bf(in[i]);
}

// ------------- f32 [K,N] -> bf16 transposed [N,K] (weights only) -----------
__global__ void cvtT_f32_bf16(const float* __restrict__ in,  // [K,N]
                              unsigned short* __restrict__ out,  // [N,K]
                              int Kd, int Nd) {
  int i = blockIdx.x * blockDim.x + threadIdx.x;
  int total = Kd * Nd;
  int stride = gridDim.x * blockDim.x;
  for (; i < total; i += stride) {
    int k = i / Nd, n = i - k * Nd;
    out[(size_t)n * Kd + k] = f2bf(in[i]);
  }
}

// ------------------------------ WMMA GEMM ----------------------------------
// C[M,N] = act(A[M,K] @ Bt[N,K]^T + bias[N])
// A bf16 row-major [M,K]; Bt bf16 row-major [N,K] (pre-transposed weight).
// Block: 256 threads = 8 waves as 4(M) x 2(N). Block tile 128 x 128, K-step 32.
// Each wave owns a 32x64 patch: 2 A-frags x 4 B-frags -> 8 WMMAs / K-step.
// Double-buffered LDS; async global->LDS staging when available.
// Requires: M%128==0, N%128==0, K%32==0 (true for every layer here).

// Stage a 128-row x 32-col bf16 tile (row stride ld) into LDS [row][k].
__device__ __forceinline__ void stage_tile(const unsigned short* __restrict__ src,
                                           int row0, int ld, int k0,
                                           unsigned short* dst, int t) {
#pragma unroll
  for (int it = 0; it < 2; ++it) {
    int q = t + it * 256;
    int r = q >> 2, c = q & 3;  // c: 8-element chunk within the 32-wide row
    const unsigned short* g = src + (size_t)(row0 + r) * ld + k0 + c * 8;
    unsigned short* l = dst + r * 32 + c * 8;
#if HAVE_ASYNC_LDS
    __builtin_amdgcn_global_load_async_to_lds_b128((gv4*)g, (lv4*)l, 0, 0);
#else
    *(uint4*)l = *(const uint4*)g;
#endif
  }
}

template <bool ACT, bool OUT_BF16>
__global__ __launch_bounds__(256) void gemm_bf16_wmma(
    const unsigned short* __restrict__ A,    // [M,K]
    const unsigned short* __restrict__ Bt,   // [N,K]
    const float* __restrict__ bias,          // [N]
    void* __restrict__ Cout,
    int M, int N, int K) {
  __shared__ unsigned short As[2][128 * 32];  // [row][k]
  __shared__ unsigned short Bs[2][128 * 32];  // [col][k]

  const int n0   = blockIdx.x * 128;
  const int m0   = blockIdx.y * 128;
  const int t    = threadIdx.x;
  const int wave = t >> 5;
  const int lane = t & 31;
  const int half = lane >> 4;  // 0: lanes 0-15, 1: lanes 16-31
  const int lmod = lane & 15;
  const int wr   = wave >> 1;  // 0..3 : M strip (32 rows)
  const int wc   = wave & 1;   // 0..1 : N strip (64 cols)

  v8f acc[2][4];
#pragma unroll
  for (int i = 0; i < 2; ++i)
#pragma unroll
    for (int j = 0; j < 4; ++j)
#pragma unroll
      for (int r = 0; r < 8; ++r) acc[i][j][r] = 0.f;

  // ---- prologue: stage K-step 0 into buffer 0 ----
  stage_tile(A, m0, K, 0, As[0], t);
  stage_tile(Bt, n0, K, 0, Bs[0], t);
  wait_async_lds();
  __syncthreads();

  const int nsteps = K >> 5;
  for (int s = 0; s < nsteps; ++s) {
    const int cur = s & 1, nxt = cur ^ 1;
    if (s + 1 < nsteps) {  // overlap next-tile staging with compute
      stage_tile(A, m0, K, (s + 1) * 32, As[nxt], t);
      stage_tile(Bt, n0, K, (s + 1) * 32, Bs[nxt], t);
    }

    // ---- fragments (contiguous k = 16*half + s in both A and B) ----
    union frag_t { unsigned int u[8]; v16bf v; };
    frag_t a[2], b[4];
#pragma unroll
    for (int i = 0; i < 2; ++i)
#pragma unroll
      for (int vv = 0; vv < 8; ++vv) {
        int grp = vv >> 2, vi = vv & 3;
        int k = half * 16 + vv * 2;  // A: linear K per half-wave
        a[i].u[vv] = *(const unsigned int*)(
            &As[cur][(wr * 32 + i * 16 + lmod) * 32 + k]);
      }
#pragma unroll
    for (int j = 0; j < 4; ++j)
#pragma unroll
      for (int vv = 0; vv < 8; ++vv) {
        int k = half * 16 + vv * 2;  // B: linear K per half-wave
        b[j].u[vv] = *(const unsigned int*)(
            &Bs[cur][(wc * 64 + j * 16 + lmod) * 32 + k]);
      }

    // ---- 8 independent WMMA accumulation chains ----
#pragma unroll
    for (int i = 0; i < 2; ++i)
#pragma unroll
      for (int j = 0; j < 4; ++j)
        acc[i][j] = __builtin_amdgcn_wmma_f32_16x16x32_bf16(
            false, a[i].v, false, b[j].v, (short)0, acc[i][j], false, false);

    wait_async_lds();
    __syncthreads();
  }

  // ---- epilogue: C/D layout -> row = base + half*8 + r, col = lmod ----
#pragma unroll
  for (int i = 0; i < 2; ++i)
#pragma unroll
    for (int j = 0; j < 4; ++j) {
      int col  = n0 + wc * 64 + j * 16 + lmod;
      float bb = bias[col];
#pragma unroll
      for (int r = 0; r < 8; ++r) {
        int row = m0 + wr * 32 + i * 16 + half * 8 + r;
        float v = acc[i][j][r] + bb;
        if (ACT) v = lrelu(v);
        if (OUT_BF16)
          ((unsigned short*)Cout)[(size_t)row * N + col] = f2bf(v);
        else
          ((float*)Cout)[(size_t)row * N + col] = v;
      }
    }
}

// ----------------------- segment sum (sorted ids) --------------------------
__global__ __launch_bounds__(256) void seg_sum_k(
    const unsigned short* __restrict__ emb,  // [N,256] bf16
    const int* __restrict__ ids,
    float* __restrict__ sums) {  // [G,256]
  int c  = threadIdx.x;
  int i0 = blockIdx.x * 64;
  int gcur = ids[i0];
  float acc = 0.f;
  for (int i = i0; i < i0 + 64; ++i) {
    int g = ids[i];
    if (g != gcur) {
      atomicAdd(&sums[(size_t)gcur * 256 + c], acc);
      acc = 0.f;
      gcur = g;
    }
    acc += bf2f(emb[(size_t)i * 256 + c]);
  }
  atomicAdd(&sums[(size_t)gcur * 256 + c], acc);
}

__global__ void count_k(const int* __restrict__ ids,
                        float* __restrict__ counts, int n) {
  int i = blockIdx.x * blockDim.x + threadIdx.x;
  if (i < n) atomicAdd(&counts[ids[i]], 1.0f);
}

__global__ void sigma_k(const float* __restrict__ sums,
                        const float* __restrict__ counts,
                        unsigned short* __restrict__ sig, int GK) {
  int i = blockIdx.x * blockDim.x + threadIdx.x;
  if (i < GK) {
    float cnt = counts[i >> 8];  // K=256 columns
    sig[i] = f2bf(sums[i] / fmaxf(cnt, 1.0f));
  }
}

// -------- p = lrelu(x_hid + sigma_hid[batch_ids[n]]) (H=512, in place) -----
__global__ void gather_add_k(const unsigned short* __restrict__ xh,
                             const unsigned short* __restrict__ sh,
                             const int* __restrict__ ids,
                             unsigned short* __restrict__ p, int total) {
  int i = blockIdx.x * blockDim.x + threadIdx.x;
  if (i < total) {
    int n = i >> 9, c = i & 511;
    int g = ids[n];
    float v = bf2f(xh[i]) + bf2f(sh[(size_t)g * 512 + c]);
    p[i] = f2bf(lrelu(v));
  }
}

// ---------------------------------------------------------------------------
extern "C" void kernel_launch(void* const* d_in, const int* in_sizes, int n_in,
                              void* d_out, int out_size, void* d_ws,
                              size_t ws_size, hipStream_t stream) {
  (void)in_sizes; (void)n_in; (void)out_size; (void)ws_size;
  const int NN = 131072, D = 128, H = 512, KK = 256, G = 1024;

  const float* x      = (const float*)d_in[0];
  const int*   ids    = (const int*)d_in[1];
  const float* ge_W0  = (const float*)d_in[2];
  const float* ge_b0  = (const float*)d_in[3];
  const float* ge_Wh  = (const float*)d_in[4];
  const float* ge_bh  = (const float*)d_in[5];
  const float* ge_Wo  = (const float*)d_in[6];
  const float* ge_bo  = (const float*)d_in[7];
  const float* in_W   = (const float*)d_in[8];
  const float* in_b   = (const float*)d_in[9];
  const float* ag_W   = (const float*)d_in[10];
  const float* ag_b   = (const float*)d_in[11];
  const float* psi_Wh = (const float*)d_in[12];
  const float* psi_bh = (const float*)d_in[13];
  const float* psi_Wo = (const float*)d_in[14];
  const float* psi_bo = (const float*)d_in[15];

  // ---- workspace carve-up, 256B aligned regions ----
  char* ws = (char*)d_ws;
  size_t off = 0;
  auto carve = [&](size_t bytes) {
    char* p = ws + off;
    off += (bytes + 255) & ~(size_t)255;
    return p;
  };
  unsigned short* xb      = (unsigned short*)carve((size_t)NN * D * 2);
  unsigned short* wGeW0T  = (unsigned short*)carve((size_t)H * D * 2);
  unsigned short* wGeWhT  = (unsigned short*)carve((size_t)2 * H * H * 2);
  unsigned short* wGeWoT  = (unsigned short*)carve((size_t)KK * H * 2);
  unsigned short* wInWT   = (unsigned short*)carve((size_t)H * D * 2);
  unsigned short* wAgWT   = (unsigned short*)carve((size_t)H * KK * 2);
  unsigned short* wPsiWhT = (unsigned short*)carve((size_t)3 * H * H * 2);
  unsigned short* wPsiWoT = (unsigned short*)carve((size_t)D * H * 2);
  unsigned short* bufA    = (unsigned short*)carve((size_t)NN * H * 2);
  unsigned short* bufB    = (unsigned short*)carve((size_t)NN * H * 2);
  float*          sums    = (float*)carve((size_t)G * KK * 4);
  float*          counts  = (float*)carve((size_t)G * 4);
  unsigned short* sigBf   = (unsigned short*)carve((size_t)G * KK * 2);
  unsigned short* sigHid  = (unsigned short*)carve((size_t)G * H * 2);

  // ---- convert x; convert + transpose all weight matrices ----
  auto cvt = [&](const float* src, unsigned short* dst, int n) {
    int blocks = (n + 255) / 256;
    if (blocks > 4096) blocks = 4096;
    cvt_f32_bf16<<<blocks, 256, 0, stream>>>(src, dst, n);
  };
  auto cvtT = [&](const float* src, unsigned short* dst, int Kd, int Nd) {
    int total = Kd * Nd;
    int blocks = (total + 255) / 256;
    if (blocks > 2048) blocks = 2048;
    cvtT_f32_bf16<<<blocks, 256, 0, stream>>>(src, dst, Kd, Nd);
  };
  cvt(x, xb, NN * D);
  cvtT(ge_W0, wGeW0T, D, H);                       // [D,H] -> [H,D]
  cvtT(ge_Wh, wGeWhT, H, H);                       // layer 0
  cvtT(ge_Wh + H * H, wGeWhT + H * H, H, H);       // layer 1
  cvtT(ge_Wo, wGeWoT, H, KK);                      // [H,K] -> [K,H]
  cvtT(in_W, wInWT, D, H);                         // [D,H] -> [H,D]
  cvtT(ag_W, wAgWT, KK, H);                        // [K,H] -> [H,K]
  cvtT(psi_Wh, wPsiWhT, H, H);
  cvtT(psi_Wh + (size_t)H * H, wPsiWhT + (size_t)H * H, H, H);
  cvtT(psi_Wh + (size_t)2 * H * H, wPsiWhT + (size_t)2 * H * H, H, H);
  cvtT(psi_Wo, wPsiWoT, H, D);                     // [H,D] -> [D,H]

  auto gemm_act_bf = [&](const unsigned short* A, const unsigned short* Bt,
                         const float* bias, void* C, int M, int N, int K) {
    dim3 grid(N / 128, M / 128);
    gemm_bf16_wmma<true, true><<<grid, 256, 0, stream>>>(A, Bt, bias, C, M, N, K);
  };
  auto gemm_lin_bf = [&](const unsigned short* A, const unsigned short* Bt,
                         const float* bias, void* C, int M, int N, int K) {
    dim3 grid(N / 128, M / 128);
    gemm_bf16_wmma<false, true><<<grid, 256, 0, stream>>>(A, Bt, bias, C, M, N, K);
  };

  // ---- GlobalEmbedding phi MLP (trailing lrelu on output) ----
  gemm_act_bf(xb,   wGeW0T,         ge_b0,     bufA, NN, H,  D);  // h0
  gemm_act_bf(bufA, wGeWhT,         ge_bh,     bufB, NN, H,  H);  // h1
  gemm_act_bf(bufB, wGeWhT + H * H, ge_bh + H, bufA, NN, H,  H);  // h2
  gemm_act_bf(bufA, wGeWoT,         ge_bo,     bufB, NN, KK, H);  // node_emb

  // ---- per-graph mean ----
  (void)hipMemsetAsync(sums, 0, (size_t)G * KK * 4, stream);
  (void)hipMemsetAsync(counts, 0, (size_t)G * 4, stream);
  seg_sum_k<<<NN / 64, 256, 0, stream>>>(bufB, ids, sums);
  count_k<<<(NN + 255) / 256, 256, 0, stream>>>(ids, counts, NN);
  sigma_k<<<(G * KK) / 256, 256, 0, stream>>>(sums, counts, sigBf, G * KK);

  // ---- sigma_hid = sigma @ ag_W + ag_b ; x_hid = x @ in_W + in_b ----
  gemm_lin_bf(sigBf, wAgWT, ag_b, sigHid, G,  H, KK);
  gemm_lin_bf(xb,    wInWT, in_b, bufA,   NN, H, D);

  // ---- p = lrelu(x_hid + sigma_hid[ids]) (in place) ----
  gather_add_k<<<(NN * H) / 256, 256, 0, stream>>>(bufA, sigHid, ids, bufA,
                                                   NN * H);

  // ---- psi MLP ----
  gemm_act_bf(bufA, wPsiWhT,                     psi_bh,         bufB, NN, H, H);
  gemm_act_bf(bufB, wPsiWhT + (size_t)H * H,     psi_bh + H,     bufA, NN, H, H);
  gemm_act_bf(bufA, wPsiWhT + (size_t)2 * H * H, psi_bh + 2 * H, bufB, NN, H, H);

  // ---- final: out = p @ psi_Wo + psi_bo (f32 output) ----
  {
    dim3 grid(D / 128, NN / 128);
    gemm_bf16_wmma<false, false><<<grid, 256, 0, stream>>>(
        bufB, wPsiWoT, psi_bo, d_out, NN, D, H);
  }
}
